// MCFBlock_52956946760393
// MI455X (gfx1250) — matrix operand order repaired and checked by reading, not compiled, not run on hardware
//
#include <hip/hip_runtime.h>

// ---------------------------------------------------------------------------
// MCF block (channel-attention branch + Mamba + Conformer conv + FF) for
// MI455X / gfx1250.  All dense layers run through a bf16 WMMA GEMM
// (v_wmma_f32_16x16x32_bf16, f32 accumulate) with contiguous-K fragment
// layout so each A/B fragment is two unguarded global_load_b128s.  The
// sequential selective scan is a wave32 VALU kernel with LDS-staged B/C rows
// and shuffle reductions.
// ---------------------------------------------------------------------------

typedef __attribute__((ext_vector_type(16))) __bf16 v16bf;
typedef __attribute__((ext_vector_type(8)))  float  v8f;

union BFrag { v16bf v; unsigned short u[16]; uint4 q[2]; };

#define DEVFN __device__ __forceinline__

DEVFN unsigned short f2bf(float f) {
  union { float f; unsigned int u; } c; c.f = f;
  unsigned int r = c.u + 0x7FFFu + ((c.u >> 16) & 1u);   // round-to-nearest-even
  return (unsigned short)(r >> 16);
}
DEVFN float sigm(float x) { return 1.0f / (1.0f + __expf(-x)); }

DEVFN v16bf load_afrag(const unsigned short* p) {
  // lane's 16 A elements: one contiguous K run [kb+half*16, +16)
  BFrag f;
  f.q[0] = *(const uint4*)(p);
  f.q[1] = *(const uint4*)(p + 8);
  return f.v;
}
DEVFN v16bf load_bfrag(const unsigned short* p) {
  // lane's 16 B elements: one contiguous K run [kb+half*16, +16)
  BFrag f;
  f.q[0] = *(const uint4*)(p);
  f.q[1] = *(const uint4*)(p + 8);
  return f.v;
}

// ---------------------------------------------------------------------------
// Generalized batched GEMM, contiguous-K operands (sAk == sBk == 1):
//   out[b,m,n] = act(sum_k A[b,m,k]*B[b,n(row),k] ... ) * alpha
//                + resScale * res[b,m,n]      (+ bias[n] before act)
// Requirements (arranged by the host): M % 16 == 0 and grid exact in M;
// K % 32 == 0; B buffer padded so rows up to ceil(N/64)*64-1 are readable.
// act: 0 none, 1 relu, 2 silu, 3 softplus
// ---------------------------------------------------------------------------
__global__ __launch_bounds__(256) void gemm_wmma_bf16(
    const unsigned short* __restrict__ A, const unsigned short* __restrict__ Bm,
    const float* __restrict__ bias, const float* __restrict__ res,
    float* __restrict__ outF, unsigned short* __restrict__ outB,
    int M, int N, int K,
    long long sAb, long long sAm,
    long long sBb, long long sBn,
    long long sOb, long long sOm, long long sOn,
    long long sRb, long long sRm, long long sRn,
    float alpha, float resScale, int act)
{
  const int lane = threadIdx.x & 31;
  const int half = lane >> 4;
  const int l16  = lane & 15;
  const int b    = blockIdx.z;
  const int m0   = blockIdx.y * 128 + (threadIdx.x >> 5) * 16;
  const int n0b  = blockIdx.x * 64;

  const unsigned short* pa  = A  + b * sAb + (long long)(m0 + l16) * sAm + half * 16;
  const unsigned short* pb0 = Bm + b * sBb + (long long)(n0b + l16) * sBn + half * 16;
  const unsigned short* pb1 = pb0 + 16 * sBn;
  const unsigned short* pb2 = pb0 + 32 * sBn;
  const unsigned short* pb3 = pb0 + 48 * sBn;

  v8f zero = {0.f,0.f,0.f,0.f,0.f,0.f,0.f,0.f};
  v8f acc[4];
#pragma unroll
  for (int t = 0; t < 4; ++t) acc[t] = zero;

#pragma unroll 2
  for (int kb = 0; kb < K; kb += 32) {
    v16bf af = load_afrag(pa);
    acc[0] = __builtin_amdgcn_wmma_f32_16x16x32_bf16(false, af, false, load_bfrag(pb0),
                                                     (short)0, acc[0], false, false);
    acc[1] = __builtin_amdgcn_wmma_f32_16x16x32_bf16(false, af, false, load_bfrag(pb1),
                                                     (short)0, acc[1], false, false);
    acc[2] = __builtin_amdgcn_wmma_f32_16x16x32_bf16(false, af, false, load_bfrag(pb2),
                                                     (short)0, acc[2], false, false);
    acc[3] = __builtin_amdgcn_wmma_f32_16x16x32_bf16(false, af, false, load_bfrag(pb3),
                                                     (short)0, acc[3], false, false);
    pa += 32; pb0 += 32; pb1 += 32; pb2 += 32; pb3 += 32;
  }

  // D layout: elem r -> (M = m0 + half*8 + r, N = n0 + (lane&15))
#pragma unroll
  for (int t = 0; t < 4; ++t) {
    int nn = n0b + t * 16 + l16;
    if (nn >= N) continue;
    float bv = bias ? bias[nn] : 0.f;
#pragma unroll
    for (int r = 0; r < 8; ++r) {
      int mm = m0 + half * 8 + r;
      float v = acc[t][r] + bv;
      if      (act == 1) v = fmaxf(v, 0.f);
      else if (act == 2) v = v * sigm(v);
      else if (act == 3) v = (v > 20.f) ? v : log1pf(__expf(v));
      v *= alpha;
      if (res) v += resScale * res[(long long)b * sRb + (long long)mm * sRm + (long long)nn * sRn];
      long long o = (long long)b * sOb + (long long)mm * sOm + (long long)nn * sOn;
      if (outF) outF[o] = v;
      if (outB) outB[o] = f2bf(v);
    }
  }
}

// ---------------------------------------------------------------------------
// fp32 -> bf16 conversion
// ---------------------------------------------------------------------------
__global__ void f32_to_bf16(const float* __restrict__ s, unsigned short* __restrict__ d, int n) {
  for (int i = blockIdx.x * blockDim.x + threadIdx.x; i < n; i += gridDim.x * blockDim.x)
    d[i] = f2bf(s[i]);
}

// fp32 -> bf16 with row/col zero-padding (strided source rows).
// dst[r*dstCols + c] = (r<srcRows && c<srcCols) ? src[r*srcStride + c] : 0
__global__ void cvt_pad(const float* __restrict__ s, unsigned short* __restrict__ d,
                        int srcRows, int srcCols, long long srcStride,
                        int dstCols, int dstRows) {
  int total = dstRows * dstCols;
  for (int i = blockIdx.x * blockDim.x + threadIdx.x; i < total; i += gridDim.x * blockDim.x) {
    int r = i / dstCols, c = i % dstCols;
    d[i] = (r < srcRows && c < srcCols) ? f2bf(s[(long long)r * srcStride + c]) : (unsigned short)0;
  }
}

// Materialize the "faithful reshape" view of c1 for the Q/K projections:
// c1r[(b*512+n)*256 + ci] = c1f[b*131072 + ci*512 + n]   (bf16)
__global__ void reshape_cab(const float* __restrict__ c1f, unsigned short* __restrict__ c1r) {
  const int total = 8 * 512 * 256;
  for (int i = blockIdx.x * blockDim.x + threadIdx.x; i < total; i += gridDim.x * blockDim.x) {
    int ci = i & 255, n = (i >> 8) & 511, b = i >> 17;
    c1r[i] = f2bf(c1f[(long long)b * 131072 + ci * 512 + n]);
  }
}

// ---------------------------------------------------------------------------
// im2col for k=3, pad=1 conv over a strided (b, ci, pos) view.
// dst[(b*512+n)*768 + ci*3 + dk] matches weight layout (O, I, 3) flattened.
// ---------------------------------------------------------------------------
__global__ void im2col3(const float* __restrict__ src, unsigned short* __restrict__ dst,
                        long long sB, long long sC, long long sP) {
  const int total = 8 * 512 * 256 * 3;
  for (int i = blockIdx.x * blockDim.x + threadIdx.x; i < total; i += gridDim.x * blockDim.x) {
    int dk = i % 3, ci = (i / 3) % 256, n = (i / 768) % 512, b = i / (768 * 512);
    int pos = n + dk - 1;
    float f = (pos >= 0 && pos < 512)
                  ? src[(long long)b * sB + (long long)ci * sC + (long long)pos * sP] : 0.f;
    dst[i] = f2bf(f);
  }
}

// ---------------------------------------------------------------------------
// LayerNorm over last dim (256), wave per row; optional relu + residual.
// ---------------------------------------------------------------------------
__global__ __launch_bounds__(256) void layernorm256(
    const float* __restrict__ src, const float* __restrict__ res,
    const float* __restrict__ g, const float* __restrict__ bt,
    float* __restrict__ outF, unsigned short* __restrict__ outB, int rows, int act)
{
  int row = blockIdx.x * 8 + (threadIdx.x >> 5);
  if (row >= rows) return;
  int lane = threadIdx.x & 31;
  const float* p = src + (long long)row * 256;
  float v[8], s = 0.f;
#pragma unroll
  for (int i = 0; i < 8; ++i) { v[i] = p[lane + 32 * i]; s += v[i]; }
#pragma unroll
  for (int off = 16; off; off >>= 1) s += __shfl_xor(s, off, 32);
  float mean = s * (1.f / 256.f), q = 0.f;
#pragma unroll
  for (int i = 0; i < 8; ++i) { float d = v[i] - mean; q += d * d; }
#pragma unroll
  for (int off = 16; off; off >>= 1) q += __shfl_xor(q, off, 32);
  float inv = rsqrtf(q * (1.f / 256.f) + 1e-5f);
#pragma unroll
  for (int i = 0; i < 8; ++i) {
    int c = lane + 32 * i;
    float o = (v[i] - mean) * inv * g[c] + bt[c];
    if (act == 1) o = fmaxf(o, 0.f);
    if (res) o += res[(long long)row * 256 + c];
    long long idx = (long long)row * 256 + c;
    if (outF) outF[idx] = o;
    if (outB) outB[idx] = f2bf(o);
  }
}

// Softmax over last dim (256), wave per row; in-place fp32 + TRANSPOSED bf16
// mirror ccT[b][j][i] so the attention-apply GEMM gets contiguous-K B.
__global__ __launch_bounds__(256) void softmax256(float* __restrict__ p,
                                                  unsigned short* __restrict__ outBT, int rows) {
  int row = blockIdx.x * 8 + (threadIdx.x >> 5);
  if (row >= rows) return;
  int lane = threadIdx.x & 31;
  float* d = p + (long long)row * 256;
  float v[8], mx = -3.0e38f;
#pragma unroll
  for (int i = 0; i < 8; ++i) { v[i] = d[lane + 32 * i]; mx = fmaxf(mx, v[i]); }
#pragma unroll
  for (int off = 16; off; off >>= 1) mx = fmaxf(mx, __shfl_xor(mx, off, 32));
  float s = 0.f;
#pragma unroll
  for (int i = 0; i < 8; ++i) { v[i] = __expf(v[i] - mx); s += v[i]; }
#pragma unroll
  for (int off = 16; off; off >>= 1) s += __shfl_xor(s, off, 32);
  float inv = 1.f / s;
  int b = row >> 8, irow = row & 255;
#pragma unroll
  for (int i = 0; i < 8; ++i) {
    int j = lane + 32 * i;
    float o = v[i] * inv;
    d[j] = o;
    outBT[(long long)b * 65536 + (long long)j * 256 + irow] = f2bf(o);
  }
}

// Mamba causal depthwise conv (k=4, pad 3 left) + SiLU.  xm = xz[:, :512].
__global__ void mamba_dwconv(const float* __restrict__ xz, const float* __restrict__ w,
                             const float* __restrict__ bc,
                             float* __restrict__ xt, unsigned short* __restrict__ xtb) {
  const int total = 8 * 512 * 512;
  for (int i = blockIdx.x * blockDim.x + threadIdx.x; i < total; i += gridDim.x * blockDim.x) {
    int d = i & 511, l = (i >> 9) & 511, b = i >> 18;
    float a = bc[d];
#pragma unroll
    for (int j = 0; j < 4; ++j) {
      int ll = l - 3 + j;
      if (ll >= 0) a += w[d * 4 + j] * xz[((long long)(b * 512 + ll)) * 1024 + d];
    }
    a = a * sigm(a);
    xt[i] = a;
    xtb[i] = f2bf(a);
  }
}

// ---------------------------------------------------------------------------
// Selective scan.  One wave per (b, d); 256 states striped 8/lane in VGPRs.
// 16 waves per block share LDS-staged B/C rows (one (b,l) row per step),
// cutting L2 traffic for B/C by 16x.  y reduction via shuffle.
// dbl row layout: [dt_in(16) | B(256) | C(256)] = 528 cols.
// ---------------------------------------------------------------------------
__global__ __launch_bounds__(512) void mamba_scan(
    const float* __restrict__ dbl, const float* __restrict__ dt,
    const float* __restrict__ xt, const float* __restrict__ A_log,
    float* __restrict__ yraw) {
  __shared__ float lds[512];                  // [0..255]=B, [256..511]=C
  const int b = blockIdx.y;
  const int wave = threadIdx.x >> 5, lane = threadIdx.x & 31;
  const int d = blockIdx.x * 16 + wave;
  float Av[8], h[8];
#pragma unroll
  for (int i = 0; i < 8; ++i) {
    int s = lane + 32 * i;
    Av[i] = -__expf(A_log[(long long)d * 256 + s]);
    h[i] = 0.f;
  }
  for (int l = 0; l < 512; ++l) {
    long long row = (long long)(b * 512 + l);
    __syncthreads();
    lds[threadIdx.x] = dbl[row * 528 + 16 + threadIdx.x];
    __syncthreads();
    if (l + 1 < 512)                          // global_prefetch next B/C row
      __builtin_prefetch(&dbl[(row + 1) * 528 + 16 + threadIdx.x], 0, 1);
    float dtv = dt[row * 512 + d];
    float xv  = xt[row * 512 + d];
    float dx  = dtv * xv;
    float part = 0.f;
#pragma unroll
    for (int i = 0; i < 8; ++i) {
      int s = lane + 32 * i;
      float dA = __expf(dtv * Av[i]);
      h[i] = h[i] * dA + dx * lds[s];
      part += h[i] * lds[256 + s];
    }
#pragma unroll
    for (int off = 16; off; off >>= 1) part += __shfl_xor(part, off, 32);
    if (lane == 0) yraw[row * 512 + d] = part;
  }
}

// y = (scan + xt*D) * silu(z);  z = xz[:, 512:]
__global__ void mamba_ypost(const float* __restrict__ yraw, const float* __restrict__ xt,
                            const float* __restrict__ Dp, const float* __restrict__ xz,
                            unsigned short* __restrict__ ybf) {
  const int total = 8 * 512 * 512;
  for (int i = blockIdx.x * blockDim.x + threadIdx.x; i < total; i += gridDim.x * blockDim.x) {
    int d = i & 511, row = i >> 9;
    float z = xz[(long long)row * 1024 + 512 + d];
    float y = (yraw[i] + xt[i] * Dp[d]) * (z * sigm(z));
    ybf[i] = f2bf(y);
  }
}

// GLU: a*sigmoid(g) from hp (rows x 1024), written in (b, ch, n) layout for
// the depthwise conv's contiguous taps.
__global__ void glu_kernel(const float* __restrict__ hp, float* __restrict__ glu) {
  const int total = 8 * 512 * 512;
  for (int i = blockIdx.x * blockDim.x + threadIdx.x; i < total; i += gridDim.x * blockDim.x) {
    int n = i & 511, ch = (i >> 9) & 511, b = i >> 18;
    long long row = (long long)(b * 512 + n) * 1024;
    float a = hp[row + ch], g = hp[row + 512 + ch];
    glu[(long long)b * 262144 + (long long)ch * 512 + n] = a * sigm(g);
  }
}

// Depthwise conv k=31 pad 15 + eval-mode BatchNorm + SiLU -> bf16 (b,n,ch)
__global__ void dwconv31(const float* __restrict__ glu, const float* __restrict__ w,
                         const float* __restrict__ wb, const float* __restrict__ bng,
                         const float* __restrict__ bnb, unsigned short* __restrict__ out) {
  const int total = 8 * 512 * 512;
  const float bninv = rsqrtf(1.0f + 1e-5f);
  for (int i = blockIdx.x * blockDim.x + threadIdx.x; i < total; i += gridDim.x * blockDim.x) {
    int n = i & 511, ch = (i >> 9) & 511, b = i >> 18;
    const float* rp = glu + (long long)b * 262144 + (long long)ch * 512;
    float s = wb[ch];
#pragma unroll
    for (int t = 0; t < 31; ++t) {
      int p = n + t - 15;
      if (p >= 0 && p < 512) s += w[ch * 31 + t] * rp[p];
    }
    float h = s * bninv * bng[ch] + bnb[ch];
    h = h * sigm(h);
    out[(long long)(b * 512 + n) * 512 + ch] = f2bf(h);
  }
}

// ---------------------------------------------------------------------------
// Host orchestration
// ---------------------------------------------------------------------------
typedef long long ll;

static inline void* wsTake(char*& p, size_t bytes) {
  void* r = (void*)p;
  p += (bytes + 255) & ~(size_t)255;
  return r;
}

static void gemm(hipStream_t st, const unsigned short* A, const unsigned short* B,
                 const float* bias, const float* res, float* outF, unsigned short* outB,
                 int M, int N, int K, int batch,
                 ll sAb, ll sAm, ll sBb, ll sBn,
                 ll sOb, ll sOm, ll sOn, ll sRb, ll sRm, ll sRn,
                 float alpha, float resScale, int act) {
  dim3 g((N + 63) / 64, (M + 127) / 128, batch);
  gemm_wmma_bf16<<<g, dim3(256), 0, st>>>(A, B, bias, res, outF, outB, M, N, K,
                                          sAb, sAm, sBb, sBn,
                                          sOb, sOm, sOn, sRb, sRm, sRn,
                                          alpha, resScale, act);
}

static void cvt(hipStream_t st, const float* s, unsigned short* d, int n) {
  f32_to_bf16<<<dim3((n + 255) / 256), dim3(256), 0, st>>>(s, d, n);
}

extern "C" void kernel_launch(void* const* d_in, const int* in_sizes, int n_in,
                              void* d_out, int out_size, void* d_ws, size_t ws_size,
                              hipStream_t stream) {
  (void)in_sizes; (void)n_in; (void)out_size; (void)ws_size;
  const float* x        = (const float*)d_in[0];
  const float* conv1_w  = (const float*)d_in[1];  const float* conv1_b = (const float*)d_in[2];
  const float* ln1_g    = (const float*)d_in[3];  const float* ln1_b   = (const float*)d_in[4];
  const float* q_w      = (const float*)d_in[5];  const float* q_b     = (const float*)d_in[6];
  const float* k_w      = (const float*)d_in[7];  const float* k_b     = (const float*)d_in[8];
  const float* conv2_w  = (const float*)d_in[9];  const float* conv2_b = (const float*)d_in[10];
  const float* ln2_g    = (const float*)d_in[11]; const float* ln2_b   = (const float*)d_in[12];
  const float* m_in_w   = (const float*)d_in[13];
  const float* m_conv_w = (const float*)d_in[14]; const float* m_conv_b= (const float*)d_in[15];
  const float* m_xp_w   = (const float*)d_in[16];
  const float* m_dt_w   = (const float*)d_in[17]; const float* m_dt_b  = (const float*)d_in[18];
  const float* m_A_log  = (const float*)d_in[19]; const float* m_D     = (const float*)d_in[20];
  const float* m_out_w  = (const float*)d_in[21];
  const float* cm_ln_g  = (const float*)d_in[22]; const float* cm_ln_b = (const float*)d_in[23];
  const float* pw1_w    = (const float*)d_in[24]; const float* pw1_b   = (const float*)d_in[25];
  const float* dw_w     = (const float*)d_in[26]; const float* dw_b    = (const float*)d_in[27];
  const float* bn_g     = (const float*)d_in[28]; const float* bn_b    = (const float*)d_in[29];
  const float* pw2_w    = (const float*)d_in[30]; const float* pw2_b   = (const float*)d_in[31];
  const float* ff_ln_g  = (const float*)d_in[32]; const float* ff_ln_b = (const float*)d_in[33];
  const float* ff_w1    = (const float*)d_in[34]; const float* ff_b1   = (const float*)d_in[35];
  const float* ff_w2    = (const float*)d_in[36]; const float* ff_b2   = (const float*)d_in[37];
  const float* po_ln_g  = (const float*)d_in[38]; const float* po_ln_b = (const float*)d_in[39];
  float* out = (float*)d_out;

  const int BL = 4096;           // 8 * 512 rows
  char* wp = (char*)d_ws;
  typedef unsigned short u16;

  // bf16 weights (wb_xp padded 528->576 rows; wb_dt padded K 16->32)
  u16* wb_c1 = (u16*)wsTake(wp, 196608 * 2);
  u16* wb_q  = (u16*)wsTake(wp, 65536 * 2);
  u16* wb_k  = (u16*)wsTake(wp, 65536 * 2);
  u16* wb_c2 = (u16*)wsTake(wp, 196608 * 2);
  u16* wb_in = (u16*)wsTake(wp, 262144 * 2);
  u16* wb_xp = (u16*)wsTake(wp, (size_t)576 * 512 * 2);
  u16* wb_dt = (u16*)wsTake(wp, (size_t)512 * 32 * 2);
  u16* wb_ou = (u16*)wsTake(wp, 131072 * 2);
  u16* wb_p1 = (u16*)wsTake(wp, 262144 * 2);
  u16* wb_p2 = (u16*)wsTake(wp, 131072 * 2);
  u16* wb_f1 = (u16*)wsTake(wp, 262144 * 2);
  u16* wb_f2 = (u16*)wsTake(wp, 262144 * 2);
  // activations
  u16*   patch = (u16*)wsTake(wp, (size_t)BL * 768 * 2);
  float* t1    = (float*)wsTake(wp, (size_t)BL * 256 * 4);
  float* c1f   = (float*)wsTake(wp, (size_t)BL * 256 * 4);
  u16*   c1bf  = (u16*)wsTake(wp, (size_t)BL * 256 * 2);
  u16*   c1r   = (u16*)wsTake(wp, (size_t)BL * 256 * 2);   // reshape view, contiguous K
  u16*   qb2   = (u16*)wsTake(wp, (size_t)BL * 256 * 2);   // (b, ch, n)
  u16*   kb2   = (u16*)wsTake(wp, (size_t)BL * 256 * 2);   // (b, ch, n)
  float* ccf   = (float*)wsTake(wp, (size_t)8 * 65536 * 4);
  u16*   ccbT  = (u16*)wsTake(wp, (size_t)8 * 65536 * 2);  // transposed (b, j, i)
  float* xca   = (float*)wsTake(wp, (size_t)BL * 256 * 4);
  float* xa    = (float*)wsTake(wp, (size_t)BL * 256 * 4);
  u16*   xabf  = (u16*)wsTake(wp, (size_t)BL * 256 * 2);
  float* xz    = (float*)wsTake(wp, (size_t)BL * 1024 * 4);
  float* xt    = (float*)wsTake(wp, (size_t)BL * 512 * 4);
  u16*   xtbf  = (u16*)wsTake(wp, (size_t)BL * 512 * 2);
  float* dbl   = (float*)wsTake(wp, (size_t)BL * 528 * 4);
  u16*   dtin  = (u16*)wsTake(wp, (size_t)BL * 32 * 2);    // packed dt_in, K padded to 32
  float* dtf   = (float*)wsTake(wp, (size_t)BL * 512 * 4);
  float* yraw  = (float*)wsTake(wp, (size_t)BL * 512 * 4);
  u16*   ybf   = (u16*)wsTake(wp, (size_t)BL * 512 * 2);
  float* xb    = (float*)wsTake(wp, (size_t)BL * 256 * 4);
  u16*   hlnbf = (u16*)wsTake(wp, (size_t)BL * 256 * 2);
  float* hp    = (float*)wsTake(wp, (size_t)BL * 1024 * 4);
  float* glu   = (float*)wsTake(wp, (size_t)BL * 512 * 4);
  u16*   hdbf  = (u16*)wsTake(wp, (size_t)BL * 512 * 2);
  float* xc    = (float*)wsTake(wp, (size_t)BL * 256 * 4);
  u16*   hffbf = (u16*)wsTake(wp, (size_t)BL * 1024 * 2);
  float* xd    = (float*)wsTake(wp, (size_t)BL * 256 * 4);

  // ---- weight conversions ----
  cvt(stream, conv1_w, wb_c1, 196608);  cvt(stream, q_w,    wb_q, 65536);
  cvt(stream, k_w,     wb_k, 65536);    cvt(stream, conv2_w, wb_c2, 196608);
  cvt(stream, m_in_w,  wb_in, 262144);  cvt(stream, m_out_w, wb_ou, 131072);
  cvt(stream, pw1_w,   wb_p1, 262144);  cvt(stream, pw2_w,  wb_p2, 131072);
  cvt(stream, ff_w1,   wb_f1, 262144);  cvt(stream, ff_w2,  wb_f2, 262144);
  cvt_pad<<<dim3(1152), dim3(256), 0, stream>>>(m_xp_w, wb_xp, 528, 512, 512, 512, 576);
  cvt_pad<<<dim3(64),   dim3(256), 0, stream>>>(m_dt_w, wb_dt, 512, 16, 16, 32, 512);

  dim3 eb(256), eg(8192);

  // ================= Phase 1: channel branch =================
  // conv1 on x viewed (faithful reshape) as (B, 256ch, 512len)
  im2col3<<<eg, eb, 0, stream>>>(x, patch, 131072, 512, 1);
  gemm(stream, patch, wb_c1, conv1_b, nullptr, t1, nullptr, BL, 256, 768, 1,
       0, 768, 0, 768, 0, 256, 1, 0, 0, 0, 1.f, 0.f, 0);
  layernorm256<<<dim3(512), dim3(256), 0, stream>>>(t1, nullptr, ln1_g, ln1_b, c1f, c1bf, BL, 1);
  // materialize reshape view for Q/K projections (contiguous K)
  reshape_cab<<<eg, eb, 0, stream>>>(c1f, c1r);
  // Q/K 1x1 convs; outputs stored (b, ch, n) so QK^T gets contiguous-K A and B
  gemm(stream, c1r, wb_q, q_b, nullptr, nullptr, qb2, 512, 256, 256, 8,
       131072, 256, 0, 256, 131072, 1, 512, 0, 0, 0, 1.f, 0.f, 0);
  gemm(stream, c1r, wb_k, k_b, nullptr, nullptr, kb2, 512, 256, 256, 8,
       131072, 256, 0, 256, 131072, 1, 512, 0, 0, 0, 1.f, 0.f, 0);
  // cc = softmax(Q @ K) over channels: batched 256x256x512, both contiguous-K
  gemm(stream, qb2, kb2, nullptr, nullptr, ccf, nullptr, 256, 256, 512, 8,
       131072, 512, 131072, 512, 65536, 256, 1, 0, 0, 0, 1.f, 0.f, 0);
  softmax256<<<dim3(256), dim3(256), 0, stream>>>(ccf, ccbT, 2048);
  // x_ca = c1 @ cc + 2*c1   (B = ccT, contiguous K)
  gemm(stream, c1bf, ccbT, nullptr, c1f, xca, nullptr, 512, 256, 256, 8,
       131072, 256, 65536, 256, 131072, 256, 1, 131072, 256, 1, 1.f, 2.f, 0);
  // conv2 on transpose(x_ca) -> (B,C,N): ci stride 1, pos stride 256
  im2col3<<<eg, eb, 0, stream>>>(xca, patch, 131072, 1, 256);
  gemm(stream, patch, wb_c2, conv2_b, nullptr, t1, nullptr, BL, 256, 768, 1,
       0, 768, 0, 768, 0, 256, 1, 0, 0, 0, 1.f, 0.f, 0);
  // xa = x + relu(LN(conv2))
  layernorm256<<<dim3(512), dim3(256), 0, stream>>>(t1, x, ln2_g, ln2_b, xa, xabf, BL, 1);

  // ================= Phase 2: Mamba =================
  gemm(stream, xabf, wb_in, nullptr, nullptr, xz, nullptr, BL, 1024, 256, 1,
       0, 256, 0, 256, 0, 1024, 1, 0, 0, 0, 1.f, 0.f, 0);
  mamba_dwconv<<<eg, eb, 0, stream>>>(xz, m_conv_w, m_conv_b, xt, xtbf);
  gemm(stream, xtbf, wb_xp, nullptr, nullptr, dbl, nullptr, BL, 528, 512, 1,
       0, 512, 0, 512, 0, 528, 1, 0, 0, 0, 1.f, 0.f, 0);
  // pack dt_in (cols 0..15 of dbl) into zero-padded 4096x32 bf16
  cvt_pad<<<dim3(512), dim3(256), 0, stream>>>(dbl, dtin, BL, 16, 528, 32, BL);
  // dt = softplus(dt_in @ dt_w^T + dt_b)
  gemm(stream, dtin, wb_dt, m_dt_b, nullptr, dtf, nullptr, BL, 512, 32, 1,
       0, 32, 0, 32, 0, 512, 1, 0, 0, 0, 1.f, 0.f, 3);
  mamba_scan<<<dim3(32, 8), dim3(512), 0, stream>>>(dbl, dtf, xt, m_A_log, yraw);
  mamba_ypost<<<eg, eb, 0, stream>>>(yraw, xt, m_D, xz, ybf);
  // xb = y @ out_w^T + xa
  gemm(stream, ybf, wb_ou, nullptr, xa, xb, nullptr, BL, 256, 512, 1,
       0, 512, 0, 512, 0, 256, 1, 0, 256, 1, 1.f, 1.f, 0);

  // ================= Phase 3: Conformer conv module =================
  layernorm256<<<dim3(512), dim3(256), 0, stream>>>(xb, nullptr, cm_ln_g, cm_ln_b,
                                                    nullptr, hlnbf, BL, 0);
  gemm(stream, hlnbf, wb_p1, pw1_b, nullptr, hp, nullptr, BL, 1024, 256, 1,
       0, 256, 0, 256, 0, 1024, 1, 0, 0, 0, 1.f, 0.f, 0);
  glu_kernel<<<eg, eb, 0, stream>>>(hp, glu);
  dwconv31<<<eg, eb, 0, stream>>>(glu, dw_w, dw_b, bn_g, bn_b, hdbf);
  gemm(stream, hdbf, wb_p2, pw2_b, xb, xc, nullptr, BL, 256, 512, 1,
       0, 512, 0, 512, 0, 256, 1, 0, 256, 1, 1.f, 1.f, 0);

  // ================= Phase 4: FF (scale 0.5) + post-LN =================
  layernorm256<<<dim3(512), dim3(256), 0, stream>>>(xc, nullptr, ff_ln_g, ff_ln_b,
                                                    nullptr, hlnbf, BL, 0);
  gemm(stream, hlnbf, wb_f1, ff_b1, nullptr, nullptr, hffbf, BL, 1024, 256, 1,
       0, 256, 0, 256, 0, 1024, 1, 0, 0, 0, 1.f, 0.f, 2);
  gemm(stream, hffbf, wb_f2, ff_b2, xc, xd, nullptr, BL, 256, 1024, 1,
       0, 1024, 0, 1024, 0, 256, 1, 0, 256, 1, 0.5f, 1.f, 0);
  layernorm256<<<dim3(512), dim3(256), 0, stream>>>(xd, nullptr, po_ln_g, po_ln_b,
                                                    out, nullptr, BL, 0);
}
